// LinearAttention_40982577938857
// MI455X (gfx1250) — matrix operand order repaired and checked by reading, not compiled, not run on hardware
//
#include <hip/hip_runtime.h>

typedef __attribute__((ext_vector_type(16))) _Float16 v16h;
typedef __attribute__((ext_vector_type(8)))  _Float16 v8h;
typedef __attribute__((ext_vector_type(8)))  float    v8f;
typedef __attribute__((ext_vector_type(4)))  unsigned int u32x4;
typedef __attribute__((ext_vector_type(8)))  int      i32x8;
typedef __attribute__((ext_vector_type(4)))  int      i32x4;

#define SEQ   2048
#define DIM   64
#define TKV   64          // kv tile length
#define WAVES 8           // waves per block
#define TQ    16          // q rows per wave
#define STR   72          // padded LDS row stride in halves (128B data + 16B pad)
#define LOG2E 1.44269504088896340736f

__device__ __forceinline__ float featmap(float x) {
    // elu(x) + 1 == (x > 0) ? x + 1 : exp(x)
    return x > 0.0f ? x + 1.0f : __expf(x);
}

__device__ __forceinline__ v8f zero8() {
    return (v8f){0.f,0.f,0.f,0.f,0.f,0.f,0.f,0.f};
}

// Max-reduce across each 16-lane group: DPP applied directly to src0 of
// v_max_num_f32 (1 VALU op per step, no LDS traffic).
__device__ __forceinline__ float rowmax16(float x) {
    float y;
    asm("v_max_num_f32_dpp %0, %1, %1 quad_perm:[1,0,3,2] row_mask:0xf bank_mask:0xf bound_ctrl:1"
        : "=v"(y) : "v"(x));
    asm("v_max_num_f32_dpp %0, %1, %1 quad_perm:[2,3,0,1] row_mask:0xf bank_mask:0xf bound_ctrl:1"
        : "=v"(x) : "v"(y));
    asm("v_max_num_f32_dpp %0, %1, %1 row_half_mirror row_mask:0xf bank_mask:0xf bound_ctrl:1"
        : "=v"(y) : "v"(x));
    asm("v_max_num_f32_dpp %0, %1, %1 row_mirror row_mask:0xf bank_mask:0xf bound_ctrl:1"
        : "=v"(x) : "v"(y));
    return x;
}

// ---- Tensor Data Mover: 2D tile (data_size=2B) global->LDS with row padding.
// group1 w0: data_size=1(2B)<<16 | pad_enable<<20 | pad_interval=4(128B)<<22
//            | pad_amount=3(4 dwords=16B)<<25  ->  stride-72-halves rows in LDS.
__device__ __forceinline__ void tdm_load_tile_2d(unsigned lds_addr, const void* gptr,
                                                 int tensor_d0, int tensor_d1,
                                                 int tile_d0, int tile_d1,
                                                 int d0_stride) {
    const unsigned long long ga = (unsigned long long)gptr;
    u32x4 g0;
    g0[0] = 1u;                                   // count=1, user descriptor
    g0[1] = lds_addr;                             // LDS byte address
    g0[2] = (unsigned)ga;                         // global_addr[31:0]
    g0[3] = (unsigned)((ga >> 32) & 0x01FFFFFFu) | 0x80000000u;  // addr[56:32] | type=2
    i32x8 g1;
    g1[0] = (1 << 16) | (1 << 20) | (4 << 22) | (3 << 25);
    g1[1] = (tensor_d0 & 0xFFFF) << 16;           // tensor_dim0[15:0]
    g1[2] = ((tensor_d0 >> 16) & 0xFFFF) | ((tensor_d1 & 0xFFFF) << 16);
    g1[3] = ((tensor_d1 >> 16) & 0xFFFF) | ((tile_d0 & 0xFFFF) << 16);
    g1[4] = (tile_d1 & 0xFFFF);                   // tile_dim1, tile_dim2=0
    g1[5] = d0_stride;                            // tensor_dim0_stride[31:0]
    g1[6] = 0;
    g1[7] = 0;
    i32x4 gz = (i32x4){0, 0, 0, 0};
#if defined(__clang_major__) && (__clang_major__ >= 23)
    i32x8 gz8 = (i32x8){0, 0, 0, 0, 0, 0, 0, 0};
    __builtin_amdgcn_tensor_load_to_lds(g0, g1, gz, gz, gz8, 0);
#else
    __builtin_amdgcn_tensor_load_to_lds(g0, g1, gz, gz, 0);
#endif
}

// ================= pre-pass: Kf16 = featmap(K); Vt16 = V^T (f16) =============
__global__ __launch_bounds__(256, 1)
void LinearAttention_prep_kernel(const float* __restrict__ K,
                                 const float* __restrict__ V,
                                 _Float16* __restrict__ Kf16,
                                 _Float16* __restrict__ Vt16) {
    __shared__ _Float16 tile[DIM][STR];
    const int tid = threadIdx.x;
    const int bh  = blockIdx.y;
    const int t0  = blockIdx.x * 64;
    const float* Kb = K + (size_t)bh * SEQ * DIM;
    const float* Vb = V + (size_t)bh * SEQ * DIM;
    _Float16*    Ko = Kf16 + (size_t)bh * SEQ * DIM;
    _Float16*    Vo = Vt16 + (size_t)bh * SEQ * DIM;
#pragma unroll
    for (int i = 0; i < 16; ++i) {
        const int idx = tid + i * 256;
        const int t = idx >> 6, d = idx & 63;
        Ko[(size_t)(t0 + t) * DIM + d] = (_Float16)featmap(Kb[(size_t)(t0 + t) * DIM + d]);
        tile[d][t] = (_Float16)Vb[(size_t)(t0 + t) * DIM + d];
    }
    __syncthreads();
#pragma unroll
    for (int i = 0; i < 16; ++i) {
        const int idx = tid + i * 256;
        const int d = idx >> 6, t = idx & 63;
        Vo[(size_t)d * SEQ + t0 + t] = tile[d][t];   // coalesced along t
    }
}

// ========================= main attention kernel =============================
__global__ __launch_bounds__(256, 1)
void LinearAttention_40982577938857_kernel(const float* __restrict__ Q,
                                           const _Float16* __restrict__ Kf16,
                                           const _Float16* __restrict__ Vt16,
                                           float* __restrict__ O) {
    __shared__ __align__(32) _Float16 sK [2][TKV * STR];       // Kf tiles  18 KB
    __shared__ __align__(32) _Float16 sVt[2][DIM * STR];       // V^T tiles 18 KB
    __shared__ __align__(32) _Float16 sP [WAVES * TQ * STR];   // P tiles   18 KB

    const int tid  = threadIdx.x;
    const int wave = tid >> 5;
    const int lane = tid & 31;
    const int l16  = lane & 15;
    const int hi   = lane >> 4;

    const int bh    = blockIdx.y;
    const int qrow0 = blockIdx.x * (WAVES * TQ) + wave * TQ;

    const float*    Qb = Q    + (size_t)bh * SEQ * DIM;
    const _Float16* Kb = Kf16 + (size_t)bh * SEQ * DIM;
    const _Float16* Vb = Vt16 + (size_t)bh * SEQ * DIM;
    float*          Ob = O    + (size_t)bh * SEQ * DIM;

    // ---- TDM prologue: tiles 0 and 1 in flight (2 descriptors per tile) ----
    if (tid < 32) {
#pragma unroll
        for (int i = 0; i < 2; ++i) {
            tdm_load_tile_2d((unsigned)(unsigned long long)&sK[i][0],
                             Kb + (size_t)i * TKV * DIM,
                             DIM, SEQ, DIM, TKV, DIM);          // 64x64, rows of d
            tdm_load_tile_2d((unsigned)(unsigned long long)&sVt[i][0],
                             Vb + (size_t)i * TKV,
                             SEQ, DIM, TKV, DIM, SEQ);          // 64x64, rows of t
        }
    }

    // ---- Qf * log2(e) as two 16x32 f16 A-fragments; softmax runs in base-2 ----
    v16h aQ[2];
    {
        const float* qrow = Qb + (size_t)(qrow0 + l16) * DIM;
#pragma unroll
        for (int c = 0; c < 2; ++c) {
            const int b0 = c * 32 + hi * 16;
#pragma unroll
            for (int j = 0; j < 8; ++j) {
                aQ[c][j]     = (_Float16)(featmap(qrow[b0 + j])      * LOG2E);
                aQ[c][8 + j] = (_Float16)(featmap(qrow[b0 + 8 + j])  * LOG2E);
            }
        }
    }

    v16h ones;   // rowsum(P) = P x ones via the matrix pipe
#pragma unroll
    for (int j = 0; j < 16; ++j) ones[j] = (_Float16)1.0f;

    v8f o[4];
#pragma unroll
    for (int i = 0; i < 4; ++i) o[i] = zero8();
    float m[8], l[8];
#pragma unroll
    for (int r = 0; r < 8; ++r) { m[r] = -3.0e38f; l[r] = 0.0f; }

    _Float16* myP = sP + wave * (TQ * STR);

    for (int t0 = 0; t0 < SEQ; t0 += TKV) {
        const int b = (t0 / TKV) & 1;
        // current tile's 2 TDM ops retired (in-order); next tile's 2 may fly
        if (tid < 32) {
            if (t0 + TKV < SEQ) __builtin_amdgcn_s_wait_tensorcnt(2);
            else                __builtin_amdgcn_s_wait_tensorcnt(0);
        }
        __syncthreads();   // publish TDM-written tiles to all waves

        // ---- S' = (Qf*log2e)(16x64) @ Kf^T(64x64): 4 N-tiles x 2 K-chunks ----
        v8f sfrag[4];
#pragma unroll
        for (int j = 0; j < 4; ++j) {
            v8f acc = zero8();
#pragma unroll
            for (int c = 0; c < 2; ++c) {
                const v16h bb = *(const v16h*)&sK[b][(j * 16 + l16) * STR + c * 32 + hi * 16];
                acc = __builtin_amdgcn_wmma_f32_16x16x32_f16(
                          false, aQ[c], false, bb, (short)0, acc, false, false);
            }
            sfrag[j] = acc;
        }

        // ---- online softmax in base-2: fused DPP row-max, exp2, O rescale ----
        float alpha[8];
#pragma unroll
        for (int r = 0; r < 8; ++r) {
            float mx = fmaxf(fmaxf(sfrag[0][r], sfrag[1][r]),
                             fmaxf(sfrag[2][r], sfrag[3][r]));
            mx = rowmax16(mx);
            const float mn = fmaxf(m[r], mx);
            alpha[r] = __builtin_amdgcn_exp2f(m[r] - mn);
            m[r] = mn;
#pragma unroll
            for (int j = 0; j < 4; ++j)
                sfrag[j][r] = __builtin_amdgcn_exp2f(sfrag[j][r] - mn);
        }
#pragma unroll
        for (int jd = 0; jd < 4; ++jd)
#pragma unroll
            for (int r = 0; r < 8; ++r)
                o[jd][r] *= alpha[r];

        // ---- C-layout f32 P -> LDS -> A-layout f16 (two 16x32 frags) ----
#pragma unroll
        for (int j = 0; j < 4; ++j)
#pragma unroll
            for (int r = 0; r < 8; ++r)
                myP[(r + 8 * hi) * STR + j * 16 + l16] = (_Float16)sfrag[j][r];
        __syncthreads();

        v16h aP[2];
        {
            const _Float16* prow = myP + l16 * STR;
#pragma unroll
            for (int c = 0; c < 2; ++c) {
                const v8h lo = *(const v8h*)(prow + c * 32 + hi * 16);
                const v8h hh = *(const v8h*)(prow + c * 32 + hi * 16 + 8);
#pragma unroll
                for (int j = 0; j < 8; ++j) { aP[c][j] = lo[j]; aP[c][8 + j] = hh[j]; }
            }
        }

        // ---- row sums via the matrix pipe: rs = P @ ones ----
        v8f rs = __builtin_amdgcn_wmma_f32_16x16x32_f16(
                     false, aP[0], false, ones, (short)0, zero8(), false, false);
        rs = __builtin_amdgcn_wmma_f32_16x16x32_f16(
                     false, aP[1], false, ones, (short)0, rs, false, false);
#pragma unroll
        for (int r = 0; r < 8; ++r) l[r] = l[r] * alpha[r] + rs[r];

        // ---- O += P(16x64) @ V(64x64): 4 N-tiles x 2 K-chunks ----
#pragma unroll
        for (int jd = 0; jd < 4; ++jd) {
#pragma unroll
            for (int c = 0; c < 2; ++c) {
                const v16h bb = *(const v16h*)&sVt[b][(jd * 16 + l16) * STR + c * 32 + hi * 16];
                o[jd] = __builtin_amdgcn_wmma_f32_16x16x32_f16(
                            false, aP[c], false, bb, (short)0, o[jd], false, false);
            }
        }

        __syncthreads();   // all waves done with buffer b before TDM refills it
        if (tid < 32 && t0 + 2 * TKV < SEQ) {
            const int tn = t0 + 2 * TKV;
            tdm_load_tile_2d((unsigned)(unsigned long long)&sK[b][0],
                             Kb + (size_t)tn * DIM, DIM, SEQ, DIM, TKV, DIM);
            tdm_load_tile_2d((unsigned)(unsigned long long)&sVt[b][0],
                             Vb + (size_t)tn, SEQ, DIM, TKV, DIM, SEQ);
        }
    }

    // ---- epilogue: O / l, coalesced f32 stores ----
#pragma unroll
    for (int r = 0; r < 8; ++r) {
        const float inv = 1.0f / l[r];
        const int row = qrow0 + r + 8 * hi;
#pragma unroll
        for (int jd = 0; jd < 4; ++jd)
            Ob[(size_t)row * DIM + jd * 16 + l16] = o[jd][r] * inv;
    }
}

extern "C" void kernel_launch(void* const* d_in, const int* in_sizes, int n_in,
                              void* d_out, int out_size, void* d_ws, size_t ws_size,
                              hipStream_t stream) {
    (void)n_in; (void)out_size; (void)ws_size;
    const float* Q = (const float*)d_in[0];
    const float* K = (const float*)d_in[1];
    const float* V = (const float*)d_in[2];
    float*       O = (float*)d_out;

    const int BH = in_sizes[0] / (SEQ * DIM);            // B*H = 64
    _Float16* Kf16 = (_Float16*)d_ws;                    // 16.8 MB
    _Float16* Vt16 = Kf16 + (size_t)BH * SEQ * DIM;      // 16.8 MB

    dim3 pgrid(SEQ / 64, BH);
    LinearAttention_prep_kernel<<<pgrid, 256, 0, stream>>>(K, V, Kf16, Vt16);

    dim3 grid(SEQ / (WAVES * TQ), BH);                   // (16, 64)
    LinearAttention_40982577938857_kernel<<<grid, 256, 0, stream>>>(Q, Kf16, Vt16, O);
}
